// SelfAttention_10187662426543
// MI455X (gfx1250) — compile-verified, matrix operand order checked by reading, not run on hardware
//
#include <hip/hip_runtime.h>
#include <hip/hip_bf16.h>

// Problem constants (from reference): B=2, S=2048, D=1024, H=16, DH=64
#define BB 2
#define SS 2048
#define DD 1024
#define HH 16
#define DHD 64
#define SCALE 0.125f

typedef __attribute__((ext_vector_type(16))) __bf16 v16bf;
typedef __attribute__((ext_vector_type(8)))  float  v8f;
typedef __attribute__((ext_vector_type(4)))  unsigned int u32x4;

// round-to-nearest-even fp32 -> bf16 bits
__device__ __forceinline__ unsigned short f2bf(float f) {
    union { float f; unsigned int u; } c; c.f = f;
    unsigned int u = c.u;
    u += 0x7FFFu + ((u >> 16) & 1u);
    return (unsigned short)(u >> 16);
}

// Load a 16x32 bf16 A-layout fragment from row-major memory (global or LDS).
// base points at (row0, k0); ld = row stride in elements (multiple of 8).
// Lane r (0-15) holds row M=r, K = hi*8 + {0..7} and K = 16 + hi*8 + {0..7}.
__device__ __forceinline__ v16bf load_frag_bf16(const unsigned short* base,
                                                int ld, int r, int hi) {
    const unsigned short* p = base + r * ld + hi * 8;
    union { v16bf v; u32x4 u[2]; } f;
    f.u[0] = *(const u32x4*)(p);
    f.u[1] = *(const u32x4*)(p + 16);
    return f.v;
}

// Same fragment layout but sourced from fp32 LDS rows (row stride SS floats),
// converting to bf16 on the fly.
__device__ __forceinline__ v16bf load_frag_ldsf(const float* base, int r, int hi) {
    const float* p = base + r * SS + hi * 8;
    union { v16bf v; unsigned short s[16]; } f;
#pragma unroll
    for (int j = 0; j < 8; ++j) {
        f.s[j]     = f2bf(p[j]);
        f.s[8 + j] = f2bf(p[16 + j]);
    }
    return f.v;
}

__device__ __forceinline__ v8f wmma_bf16(v16bf a, v16bf b, v8f c) {
    return __builtin_amdgcn_wmma_f32_16x16x32_bf16(false, a, false, b,
                                                   (short)0, c, false, false);
}

// Async copy 16 bytes global -> LDS (CDNA5 ASYNCcnt path).
// laddr: LDS byte address (low 32 bits of generic shared pointer),
// gaddr: 64-bit global address.
__device__ __forceinline__ void async_copy_b128(unsigned laddr,
                                                unsigned long long gaddr) {
    asm volatile("global_load_async_to_lds_b128 %0, %1, off"
                 :: "v"(laddr), "v"(gaddr) : "memory");
}
__device__ __forceinline__ void wait_asynccnt0() {
    asm volatile("s_wait_asynccnt 0x0" ::: "memory");
}

// ---------------------------------------------------------------------------
// fp32 -> bf16 elementwise conversion
// ---------------------------------------------------------------------------
__global__ __launch_bounds__(256)
void cvt_f32_bf16_kernel(const float* __restrict__ src,
                         unsigned short* __restrict__ dst, int n) {
    int i = blockIdx.x * 256 + threadIdx.x;
    if (i < n) dst[i] = f2bf(src[i]);
}

// ---------------------------------------------------------------------------
// GEMM: C[m,n] = sum_k A[m,k] * W[n,k] + bias[n]
// A: [4096, 1024] bf16 row-major, W: [1024, 1024] bf16 row-major.
// The 16x1024 A tile (contiguous 32 KB) is staged once per block into LDS via
// async-to-LDS DMA; all 8 waves then read fragments from LDS (8x less VMEM).
// MODE 0: bf16 row-major output [4096,1024]            (Q, K)
// MODE 1: bf16 transposed-per-head output Vt[b,h,d,s]  (V)
// MODE 2: fp32 row-major output                        (final out)
// Block = 256 thr (8 waves). Each wave: one 16-row tile x 64 cols (4 WMMA tiles).
// Grid: (M/16, N/512)
// ---------------------------------------------------------------------------
template <int MODE>
__global__ __launch_bounds__(256)
void gemm_bf16_kernel(const unsigned short* __restrict__ A,
                      const unsigned short* __restrict__ W,
                      const float* __restrict__ bias,
                      void* __restrict__ outp) {
    __shared__ unsigned short sA[16 * DD];  // 32 KB staged A tile

    const int lane = threadIdx.x & 31;
    const int wave = threadIdx.x >> 5;
    const int r  = lane & 15;
    const int hi = lane >> 4;
    const int m0 = blockIdx.x * 16;
    const int n0 = blockIdx.y * 512 + wave * 64;

    // ---- async DMA: stage contiguous 32 KB A tile into LDS ----
    {
        const unsigned short* gsrc = A + (size_t)m0 * DD;
#pragma unroll
        for (int i = 0; i < 8; ++i) {
            const int off = (threadIdx.x + i * 256) * 8;  // ushort units, 16B/thread
            async_copy_b128((unsigned)(unsigned long long)(sA + off),
                            (unsigned long long)(gsrc + off));
        }
        wait_asynccnt0();
    }
    __syncthreads();

    v8f acc[4] = {};
    for (int k0 = 0; k0 < DD; k0 += 32) {
        v16bf afrag = load_frag_bf16(sA + k0, DD, r, hi);  // ds_load path
#pragma unroll
        for (int j = 0; j < 4; ++j) {
            const unsigned short* wrow = W + (size_t)(n0 + j * 16) * DD + k0;
            if (k0 + 32 < DD) __builtin_prefetch(wrow + 32, 0, 0);
            v16bf bfrag = load_frag_bf16(wrow, DD, r, hi);
            acc[j] = wmma_bf16(afrag, bfrag, acc[j]);
        }
    }

    const int nn = lane & 15;
    const int mb = hi * 8;
#pragma unroll
    for (int j = 0; j < 4; ++j) {
        const int ng = n0 + j * 16 + nn;
        const float bv = bias[ng];
#pragma unroll
        for (int v = 0; v < 8; ++v) {
            const int mg = m0 + mb + v;
            const float val = acc[j][v] + bv;
            if (MODE == 0) {
                ((unsigned short*)outp)[(size_t)mg * DD + ng] = f2bf(val);
            } else if (MODE == 1) {
                const int b = mg >> 11, s = mg & (SS - 1);
                const int h = ng >> 6,  d = ng & (DHD - 1);
                ((unsigned short*)outp)[(((size_t)(b * HH + h) * DHD) + d) * SS + s] = f2bf(val);
            } else {
                ((float*)outp)[(size_t)mg * DD + ng] = val;
            }
        }
    }
}

// ---------------------------------------------------------------------------
// Fused attention: per block = (q-tile of 16 rows, head h, batch b).
// 256 threads = 8 waves. LDS holds the full 16 x 2048 fp32 score rows (128 KB
// -> 2 blocks per 320 KB WGP).
// Stage 1: scores = Q K^T * scale, masked  (WMMA, waves split 128 col-tiles)
// Stage 2: row softmax (16 lanes/row, shuffle reductions), write attn to d_out
// Stage 3: O = P V   (WMMA, waves split K=2048, cross-wave LDS reduction)
// ---------------------------------------------------------------------------
__global__ __launch_bounds__(256)
void attn_fused_kernel(const unsigned short* __restrict__ Qb,
                       const unsigned short* __restrict__ Kb,
                       const unsigned short* __restrict__ Vtb,
                       const int* __restrict__ mask,
                       float* __restrict__ attn_out,
                       unsigned short* __restrict__ Ob) {
    extern __shared__ float lds[];  // 16 * 2048 floats = 128 KB
    const int lane = threadIdx.x & 31;
    const int wave = threadIdx.x >> 5;
    const int r  = lane & 15;
    const int hi = lane >> 4;
    const int nn = lane & 15;
    const int mb = hi * 8;

    const int qt = blockIdx.x;   // 0..127
    const int h  = blockIdx.y;   // 0..15
    const int b  = blockIdx.z;   // 0..1
    const int q0 = qt * 16;

    // ---- Stage 1: score tiles ----
    const unsigned short* Qbase = Qb + (size_t)(b * SS + q0) * DD + h * DHD;
    const v16bf qa0 = load_frag_bf16(Qbase,      DD, r, hi);
    const v16bf qa1 = load_frag_bf16(Qbase + 32, DD, r, hi);

    for (int kt = wave; kt < SS / 16; kt += 8) {
        const unsigned short* Kbase = Kb + (size_t)(b * SS + kt * 16) * DD + h * DHD;
        if (kt + 8 < SS / 16)  // prefetch next K tile for this wave
            __builtin_prefetch(Kbase + 8 * 16 * DD, 0, 0);
        v16bf kb0 = load_frag_bf16(Kbase,      DD, r, hi);
        v16bf kb1 = load_frag_bf16(Kbase + 32, DD, r, hi);
        v8f acc = {};
        acc = wmma_bf16(qa0, kb0, acc);
        acc = wmma_bf16(qa1, kb1, acc);
#pragma unroll
        for (int v = 0; v < 8; ++v) {
            const int m   = mb + v;
            const int col = kt * 16 + nn;
            float sc = acc[v] * SCALE;
            const int mval = mask[(size_t)(b * SS + q0 + m) * SS + col];
            if (mval == 0) sc = -1e9f;
            lds[m * SS + col] = sc;
        }
    }
    __syncthreads();

    // ---- Stage 2: softmax over each row (16 lanes per row) ----
    {
        const int row = threadIdx.x >> 4;  // 0..15
        const int j   = threadIdx.x & 15;
        float mx = -3.402823466e38f;
        for (int c = j; c < SS; c += 16) mx = fmaxf(mx, lds[row * SS + c]);
#pragma unroll
        for (int d = 1; d < 16; d <<= 1) mx = fmaxf(mx, __shfl_xor(mx, d, 32));
        float sum = 0.f;
        for (int c = j; c < SS; c += 16) {
            float e = __expf(lds[row * SS + c] - mx);
            lds[row * SS + c] = e;
            sum += e;
        }
#pragma unroll
        for (int d = 1; d < 16; d <<= 1) sum += __shfl_xor(sum, d, 32);
        const float inv = 1.0f / sum;
        float* arow = attn_out + (size_t)(b * HH + h) * SS * SS + (size_t)(q0 + row) * SS;
        for (int c = j; c < SS; c += 16) {
            float p = lds[row * SS + c] * inv;
            lds[row * SS + c] = p;
            arow[c] = p;   // coalesced attn write
        }
    }
    __syncthreads();

    // ---- Stage 3: O = P @ V (per-wave K chunk of 256, then LDS reduce) ----
    v8f oacc[4] = {};
    const unsigned short* Vbase = Vtb + (size_t)(b * HH + h) * DHD * SS;
    for (int ks = 0; ks < 8; ++ks) {
        const int k0 = wave * 256 + ks * 32;
        v16bf pa = load_frag_ldsf(lds + k0, r, hi);
#pragma unroll
        for (int j = 0; j < 4; ++j) {
            v16bf vb = load_frag_bf16(Vbase + (size_t)(j * 16) * SS + k0, SS, r, hi);
            oacc[j] = wmma_bf16(pa, vb, oacc[j]);
        }
    }
    __syncthreads();  // all P reads complete before reusing LDS
#pragma unroll
    for (int j = 0; j < 4; ++j)
#pragma unroll
        for (int v = 0; v < 8; ++v)
            lds[(wave * 16 + mb + v) * DHD + j * 16 + nn] = oacc[j][v];
    __syncthreads();
    for (int e = threadIdx.x; e < 16 * DHD; e += 256) {
        const int m = e >> 6, n = e & (DHD - 1);
        float s = 0.f;
#pragma unroll
        for (int w = 0; w < 8; ++w) s += lds[(w * 16 + m) * DHD + n];
        Ob[(size_t)(b * SS + q0 + m) * DD + h * DHD + n] = f2bf(s);
    }
}

// ---------------------------------------------------------------------------
// Host launch
// ---------------------------------------------------------------------------
extern "C" void kernel_launch(void* const* d_in, const int* in_sizes, int n_in,
                              void* d_out, int out_size, void* d_ws, size_t ws_size,
                              hipStream_t stream) {
    const float* x    = (const float*)d_in[0];
    const int*   mask = (const int*)d_in[1];
    const float* wq_w = (const float*)d_in[2];
    const float* wq_b = (const float*)d_in[3];
    const float* wk_w = (const float*)d_in[4];
    const float* wk_b = (const float*)d_in[5];
    const float* wv_w = (const float*)d_in[6];
    const float* wv_b = (const float*)d_in[7];
    const float* wo_w = (const float*)d_in[8];
    const float* wo_b = (const float*)d_in[9];

    // workspace layout (bytes)
    unsigned char* ws = (unsigned char*)d_ws;
    const size_t MB = 1u << 20;
    unsigned short* xb  = (unsigned short*)(ws + 0);        // 8 MB  x bf16
    unsigned short* wqb = (unsigned short*)(ws + 8  * MB);  // 2 MB
    unsigned short* wkb = (unsigned short*)(ws + 10 * MB);  // 2 MB
    unsigned short* wvb = (unsigned short*)(ws + 12 * MB);  // 2 MB
    unsigned short* wob = (unsigned short*)(ws + 14 * MB);  // 2 MB
    unsigned short* Qb  = (unsigned short*)(ws + 16 * MB);  // 8 MB
    unsigned short* Kb  = (unsigned short*)(ws + 24 * MB);  // 8 MB
    unsigned short* Vtb = (unsigned short*)(ws + 32 * MB);  // 8 MB  V^T per head
    unsigned short* Ob  = (unsigned short*)(ws + 40 * MB);  // 8 MB

    const int nx = BB * SS * DD;   // 4194304
    const int nw = DD * DD;        // 1048576

    cvt_f32_bf16_kernel<<<(nx + 255) / 256, 256, 0, stream>>>(x, xb, nx);
    cvt_f32_bf16_kernel<<<(nw + 255) / 256, 256, 0, stream>>>(wq_w, wqb, nw);
    cvt_f32_bf16_kernel<<<(nw + 255) / 256, 256, 0, stream>>>(wk_w, wkb, nw);
    cvt_f32_bf16_kernel<<<(nw + 255) / 256, 256, 0, stream>>>(wv_w, wvb, nw);
    cvt_f32_bf16_kernel<<<(nw + 255) / 256, 256, 0, stream>>>(wo_w, wob, nw);

    // Q/K/V projections (M=4096 rows, N=1024, K=1024)
    dim3 ggrid(BB * SS / 16, DD / 512);
    gemm_bf16_kernel<0><<<ggrid, 256, 0, stream>>>(xb, wqb, wq_b, Qb);
    gemm_bf16_kernel<0><<<ggrid, 256, 0, stream>>>(xb, wkb, wk_b, Kb);
    gemm_bf16_kernel<1><<<ggrid, 256, 0, stream>>>(xb, wvb, wv_b, Vtb);

    float* out  = (float*)d_out;
    float* attn = out + (size_t)BB * SS * DD;

    // fused attention: 128 KB dynamic LDS per workgroup
    static const int kLds = 16 * SS * (int)sizeof(float);  // 131072
    hipFuncSetAttribute((const void*)attn_fused_kernel,
                        hipFuncAttributeMaxDynamicSharedMemorySize, kLds);
    attn_fused_kernel<<<dim3(SS / 16, HH, BB), 256, kLds, stream>>>(
        Qb, Kb, Vtb, mask, attn, Ob);

    // output projection -> fp32 out
    gemm_bf16_kernel<2><<<ggrid, 256, 0, stream>>>(Ob, wob, wo_b, out);
}
